// MessagePassingLayer_18906446037214
// MI455X (gfx1250) — matrix operand order repaired and checked by reading, not compiled, not run on hardware
//
#include <hip/hip_runtime.h>
#include <hip/hip_bf16.h>
#include <stdint.h>

#define NN        50000
#define NE        800000
#define ND        128
#define ED        64
#define HID       128
#define KMSG      320      // 2*ND + ED
#define KUPD      256      // ND + HID
#define WAVES     4
#define LN_EPS    1e-5f

typedef __attribute__((ext_vector_type(16))) _Float16 v16h;
typedef __attribute__((ext_vector_type(8)))  _Float16 v8h;
typedef __attribute__((ext_vector_type(4)))  _Float16 v4h;
typedef __attribute__((ext_vector_type(2)))  _Float16 v2h;
typedef __attribute__((ext_vector_type(8)))  float    v8f;

union Frag16 { v16h v; struct { v8h lo, hi; } h; };

// A-fragment (16-bit A 16x32 layout): lane holds M=lane%16; halves 0..7 are
// K = grp*8+0..7, halves 8..15 are K = 16+grp*8+0..7 (grp = lane>>4).
// base must already point at row*stride + ks*32 + grp*8.
__device__ __forceinline__ v16h load_frag_a(const _Float16* base) {
  Frag16 f;
  f.h.lo = *(const v8h*)(base);
  f.h.hi = *(const v8h*)(base + 16);
  return f.v;
}

// B-fragment from pre-packed buffer: 16 contiguous halves per lane.
__device__ __forceinline__ v16h load_frag_b(const _Float16* base) {
  Frag16 f;
  f.h.lo = *(const v8h*)(base);
  f.h.hi = *(const v8h*)(base + 8);
  return f.v;
}

__device__ __forceinline__ v4h cvt4(float4 v) {
  v4h h = {(_Float16)v.x, (_Float16)v.y, (_Float16)v.z, (_Float16)v.w};
  return h;
}

// ---------------------------------------------------------------- prep ------
__global__ void zero_f32_kernel(float* __restrict__ p, int n) {
  int i = blockIdx.x * blockDim.x + threadIdx.x;
  if (i < n) p[i] = 0.0f;
}

// Pack a (Ktot x 128) f32 weight matrix into WMMA B-fragment order:
// frag[(((nt*nK)+ks)*32 + lane)*16 + i] = W[(ks*32 + (i>>3)*16 + (lane>>4)*8 + (i&7))*128 + nt*16 + (lane&15)]
__global__ void pack_b_kernel(const float* __restrict__ W, _Float16* __restrict__ frag, int nK) {
  int total = 8 * nK * 512;
  for (int t = blockIdx.x * blockDim.x + threadIdx.x; t < total; t += gridDim.x * blockDim.x) {
    int i    = t & 15;
    int lane = (t >> 4) & 31;
    int ks   = (t >> 9) % nK;
    int nt   = t / (nK << 9);
    int k = ks * 32 + (i >> 3) * 16 + (lane >> 4) * 8 + (i & 7);
    int n = nt * 16 + (lane & 15);
    frag[t] = (_Float16)W[k * 128 + n];
  }
}

// ---------------------------------------------------------------- edges -----
__global__ __launch_bounds__(128) void edge_msg_kernel(
    const float* __restrict__ nf, const long long* __restrict__ ei,
    const float* __restrict__ ef, const float* __restrict__ b1,
    const float* __restrict__ b2, const _Float16* __restrict__ w1f,
    const _Float16* __restrict__ w2f, float* __restrict__ agg,
    float* __restrict__ counts) {
  __shared__ __align__(16) _Float16 sA[WAVES][16 * KMSG];   // 40 KB: msg_in f16
  __shared__ __align__(16) _Float16 sH[WAVES][16 * HID];    // 16 KB: hidden f16
  __shared__ int sSrc[WAVES][16];
  __shared__ int sTgt[WAVES][16];

  const int wave = threadIdx.x >> 5;
  const int lane = threadIdx.x & 31;
  const int grp  = lane >> 4;
  const int col  = lane & 15;
  const int tile = blockIdx.x * WAVES + wave;   // 50000 tiles, exact
  const int e0   = tile * 16;

  if (lane < 16) {
    sSrc[wave][lane] = (int)ei[e0 + lane];
    sTgt[wave][lane] = (int)ei[NE + e0 + lane];
  }
  __syncthreads();

  // Stage msg_in = [nf[src] | nf[tgt] | ef] for 16 edges, vectorized f32->f16.
  _Float16* Ain = sA[wave];
  for (int r = 0; r < 16; ++r) {
    const long s = sSrc[wave][r];
    const long t = sTgt[wave][r];
    float4 vs = *(const float4*)&nf[s * ND + lane * 4];
    float4 vt = *(const float4*)&nf[t * ND + lane * 4];
    float2 ve = *(const float2*)&ef[(long)(e0 + r) * ED + lane * 2];
    *(v4h*)&Ain[r * KMSG + lane * 4]          = cvt4(vs);
    *(v4h*)&Ain[r * KMSG + ND + lane * 4]     = cvt4(vt);
    v2h he = {(_Float16)ve.x, (_Float16)ve.y};
    *(v2h*)&Ain[r * KMSG + 2 * ND + lane * 2] = he;
  }
  if (lane < 16) atomicAdd(&counts[sTgt[wave][lane]], 1.0f);
  __syncthreads();

  // GEMM1: (16 x 320) @ W1(320 x 128) + b1, ReLU -> sH
  _Float16* Hb = sH[wave];
  for (int nt = 0; nt < 8; ++nt) {
    float bv = b1[nt * 16 + col];
    v8f acc = {bv, bv, bv, bv, bv, bv, bv, bv};
    for (int ks = 0; ks < 10; ++ks) {
      v16h a = load_frag_a(&Ain[col * KMSG + ks * 32 + grp * 8]);
      v16h b = load_frag_b(&w1f[((nt * 10 + ks) * 32 + lane) * 16]);
      acc = __builtin_amdgcn_wmma_f32_16x16x32_f16(false, a, false, b,
                                                   (short)0, acc, false, false);
    }
    const int Mb = grp * 8;
#pragma unroll
    for (int v = 0; v < 8; ++v) {
      float x = acc[v];
      x = x > 0.0f ? x : 0.0f;
      Hb[(Mb + v) * HID + nt * 16 + col] = (_Float16)x;
    }
  }
  __syncthreads();

  // GEMM2: (16 x 128) @ W2(128 x 128) + b2 -> atomic scatter into agg[tgt]
  for (int nt = 0; nt < 8; ++nt) {
    float bv = b2[nt * 16 + col];
    v8f acc = {bv, bv, bv, bv, bv, bv, bv, bv};
    for (int ks = 0; ks < 4; ++ks) {
      v16h a = load_frag_a(&Hb[col * HID + ks * 32 + grp * 8]);
      v16h b = load_frag_b(&w2f[((nt * 4 + ks) * 32 + lane) * 16]);
      acc = __builtin_amdgcn_wmma_f32_16x16x32_f16(false, a, false, b,
                                                   (short)0, acc, false, false);
    }
    const int Mb = grp * 8;
#pragma unroll
    for (int v = 0; v < 8; ++v) {
      int t = sTgt[wave][Mb + v];
      atomicAdd(&agg[(long)t * HID + nt * 16 + col], acc[v]);
    }
  }
}

// ---------------------------------------------------------------- nodes -----
__global__ __launch_bounds__(128) void node_update_kernel(
    const float* __restrict__ nf, const float* __restrict__ agg,
    const float* __restrict__ counts, const float* __restrict__ c1v,
    const float* __restrict__ c2v, const _Float16* __restrict__ u1f,
    const _Float16* __restrict__ u2f, const float* __restrict__ gamma,
    const float* __restrict__ beta, float* __restrict__ out) {
  // sBuf holds upd_in f16 (16x256 = 8 KB) during GEMM1, then x f32 (16x128 = 8 KB).
  __shared__ __align__(16) char     sBuf[WAVES][16 * KUPD * 2];  // 32 KB
  __shared__ __align__(16) _Float16 sH[WAVES][16 * HID];         // 16 KB

  const int wave = threadIdx.x >> 5;
  const int lane = threadIdx.x & 31;
  const int grp  = lane >> 4;
  const int col  = lane & 15;
  const int tile = blockIdx.x * WAVES + wave;
  const bool valid = tile < (NN / 16);          // 3125 tiles
  const int n0 = valid ? tile * 16 : 0;

  _Float16* Ain = (_Float16*)sBuf[wave];
  for (int r = 0; r < 16; ++r) {
    const long n = n0 + r;
    float cnt = counts[n];
    cnt = cnt < 1.0f ? 1.0f : cnt;
    const float rinv = 1.0f / cnt;
    float4 vn = *(const float4*)&nf[n * ND + lane * 4];
    float4 va = *(const float4*)&agg[n * HID + lane * 4];
    va.x *= rinv; va.y *= rinv; va.z *= rinv; va.w *= rinv;
    *(v4h*)&Ain[r * KUPD + lane * 4]      = cvt4(vn);
    *(v4h*)&Ain[r * KUPD + ND + lane * 4] = cvt4(va);
  }
  __syncthreads();

  // GEMM1: (16 x 256) @ U1(256 x 128) + c1, ReLU -> sH
  _Float16* Hb = sH[wave];
  for (int nt = 0; nt < 8; ++nt) {
    float bv = c1v[nt * 16 + col];
    v8f acc = {bv, bv, bv, bv, bv, bv, bv, bv};
    for (int ks = 0; ks < 8; ++ks) {
      v16h a = load_frag_a(&Ain[col * KUPD + ks * 32 + grp * 8]);
      v16h b = load_frag_b(&u1f[((nt * 8 + ks) * 32 + lane) * 16]);
      acc = __builtin_amdgcn_wmma_f32_16x16x32_f16(false, a, false, b,
                                                   (short)0, acc, false, false);
    }
    const int Mb = grp * 8;
#pragma unroll
    for (int v = 0; v < 8; ++v) {
      float x = acc[v];
      x = x > 0.0f ? x : 0.0f;
      Hb[(Mb + v) * HID + nt * 16 + col] = (_Float16)x;
    }
  }
  __syncthreads();

  // GEMM2: (16 x 128) @ U2(128 x 128) + c2 + residual -> x (f32, reuse sBuf)
  float* Xb = (float*)sBuf[wave];
  for (int nt = 0; nt < 8; ++nt) {
    float bv = c2v[nt * 16 + col];
    v8f acc = {bv, bv, bv, bv, bv, bv, bv, bv};
    for (int ks = 0; ks < 4; ++ks) {
      v16h a = load_frag_a(&Hb[col * HID + ks * 32 + grp * 8]);
      v16h b = load_frag_b(&u2f[((nt * 4 + ks) * 32 + lane) * 16]);
      acc = __builtin_amdgcn_wmma_f32_16x16x32_f16(false, a, false, b,
                                                   (short)0, acc, false, false);
    }
    const int Mb = grp * 8;
#pragma unroll
    for (int v = 0; v < 8; ++v) {
      long n = n0 + Mb + v;
      Xb[(Mb + v) * ND + nt * 16 + col] = acc[v] + nf[n * ND + nt * 16 + col];
    }
  }
  __syncthreads();

  // Row-wise layernorm; one row per lane (lanes 0..15), float4 vectorized.
  if (lane < 16) {
    const int r = lane;
    const long n = n0 + r;
    float s = 0.0f, ss = 0.0f;
    for (int c = 0; c < ND; c += 4) {
      float4 x = *(const float4*)&Xb[r * ND + c];
      s  += x.x + x.y + x.z + x.w;
      ss += x.x * x.x + x.y * x.y + x.z * x.z + x.w * x.w;
    }
    float mu   = s * (1.0f / ND);
    float var  = ss * (1.0f / ND) - mu * mu;
    float rstd = rsqrtf(var + LN_EPS);
    if (valid) {
      for (int c = 0; c < ND; c += 4) {
        float4 x = *(const float4*)&Xb[r * ND + c];
        float4 g = *(const float4*)&gamma[c];
        float4 bb = *(const float4*)&beta[c];
        float4 o;
        o.x = (x.x - mu) * rstd * g.x + bb.x;
        o.y = (x.y - mu) * rstd * g.y + bb.y;
        o.z = (x.z - mu) * rstd * g.z + bb.z;
        o.w = (x.w - mu) * rstd * g.w + bb.w;
        *(float4*)&out[n * ND + c] = o;
      }
    }
  }
}

// ---------------------------------------------------------------- launch ----
extern "C" void kernel_launch(void* const* d_in, const int* in_sizes, int n_in,
                              void* d_out, int out_size, void* d_ws, size_t ws_size,
                              hipStream_t stream) {
  const float*     nf  = (const float*)d_in[0];
  const long long* ei  = (const long long*)d_in[1];   // int64 in reference
  const float*     ef  = (const float*)d_in[2];
  const float*     W1  = (const float*)d_in[3];
  const float*     b1  = (const float*)d_in[4];
  const float*     W2  = (const float*)d_in[5];
  const float*     b2  = (const float*)d_in[6];
  const float*     U1  = (const float*)d_in[7];
  const float*     c1  = (const float*)d_in[8];
  const float*     U2  = (const float*)d_in[9];
  const float*     c2  = (const float*)d_in[10];
  const float*     gam = (const float*)d_in[11];
  const float*     bet = (const float*)d_in[12];
  float* out = (float*)d_out;

  // workspace layout
  float*    agg    = (float*)d_ws;                 // NN*HID
  float*    counts = agg + (size_t)NN * HID;       // NN
  _Float16* w1f    = (_Float16*)(counts + NN);     // 8*10*512
  _Float16* w2f    = w1f + 8 * 10 * 512;           // 8*4*512
  _Float16* u1f    = w2f + 8 * 4 * 512;            // 8*8*512
  _Float16* u2f    = u1f + 8 * 8 * 512;            // 8*4*512

  // 1) zero accumulators
  {
    int n = NN * HID + NN;
    zero_f32_kernel<<<(n + 255) / 256, 256, 0, stream>>>(agg, n);
  }
  // 2) pack weights into WMMA B-fragment order (f16)
  pack_b_kernel<<<160, 256, 0, stream>>>(W1, w1f, 10);
  pack_b_kernel<<< 64, 256, 0, stream>>>(W2, w2f, 4);
  pack_b_kernel<<<128, 256, 0, stream>>>(U1, u1f, 8);
  pack_b_kernel<<< 64, 256, 0, stream>>>(U2, u2f, 4);
  // 3) per-edge message MLP + scatter (NE/16 = 50000 wave-tiles, 4 per block)
  edge_msg_kernel<<<NE / 16 / WAVES, 128, 0, stream>>>(nf, ei, ef, b1, b2, w1f,
                                                       w2f, agg, counts);
  // 4) node update MLP + residual layernorm (3125 wave-tiles)
  {
    int tiles = NN / 16;
    int blocks = (tiles + WAVES - 1) / WAVES;
    node_update_kernel<<<blocks, 128, 0, stream>>>(nf, agg, counts, c1, c2, u1f,
                                                   u2f, gam, bet, out);
  }
  (void)in_sizes; (void)n_in; (void)out_size; (void)ws_size;
}
